// HeteroConv_43044162240973
// MI455X (gfx1250) — matrix operand order repaired and checked by reading, not compiled, not run on hardware
//
#include <hip/hip_runtime.h>
#include <hip/hip_bf16.h>

// ---------------- problem constants (match reference) ----------------
#define NN 100000   // nodes
#define NE 200000   // edges per etype
#define NT 3        // edge types
#define DD 128      // feature dim
#define NL 3        // layers
#define BN_EPS 1e-5f

#define LDS_STRIDE 136   // 128 + 8 pad -> row stride 68 dwords: B-frag lanes hit disjoint banks

typedef __bf16 bf16;
typedef __attribute__((ext_vector_type(4)))  __bf16 v4bf;
typedef __attribute__((ext_vector_type(8)))  __bf16 v8bf;
typedef __attribute__((ext_vector_type(16))) __bf16 v16bf;
typedef __attribute__((ext_vector_type(8)))  float  v8f;
typedef __attribute__((ext_vector_type(4)))  float  v4f;

// ---------------- utility kernels ----------------

__global__ void k_zero_f32(float* __restrict__ p, int n) {
    int i = blockIdx.x * blockDim.x + threadIdx.x;
    if (i < n) p[i] = 0.0f;
}

__global__ void k_f32_to_bf16(const float* __restrict__ x, bf16* __restrict__ y, int n) {
    int i = blockIdx.x * blockDim.x + threadIdx.x;
    if (i < n) y[i] = (bf16)x[i];
}

// Convert all layer/etype weights fp32 [k][n] -> bf16 transposed [n][k]
// (transposed layout makes WMMA B-fragments two contiguous 16B reads)
__global__ void k_conv_weights(const float* __restrict__ Ws, const float* __restrict__ Wn,
                               bf16* __restrict__ WtS, bf16* __restrict__ WtN, int n) {
    int i = blockIdx.x * blockDim.x + threadIdx.x;   // over NL*NT*DD*DD
    if (i >= n) return;
    int mat = i >> 14;             // /(128*128)
    int rem = i & 16383;
    int k = rem >> 7;              // input-dim index
    int c = rem & 127;             // output-dim index
    int o = (mat << 14) + c * DD + k;
    WtS[o] = (bf16)Ws[i];
    WtN[o] = (bf16)Wn[i];
}

// per-etype in-degree counts (float, for mean divide); fixed across layers
__global__ void k_count(const int* __restrict__ dst_all, float* __restrict__ cnt, int n) {
    int i = blockIdx.x * blockDim.x + threadIdx.x;   // over NT*NE
    if (i >= n) return;
    int e = i / NE;
    atomicAdd(&cnt[e * NN + dst_all[i]], 1.0f);
}

// gather bf16 h[src] rows, scatter-add (h/deg[dst]) in fp32 into msum[dst]
// -> the scatter IS the mean aggregation; no separate normalize pass.
// 32 threads per edge, 4 cols each.
__global__ void k_scatter(const bf16* __restrict__ hb, const int* __restrict__ src,
                          const int* __restrict__ dst, const float* __restrict__ cnt_e,
                          float* __restrict__ msum, int n_threads) {
    int t = blockIdx.x * blockDim.x + threadIdx.x;
    if (t >= n_threads) return;
    int edge = t >> 5;
    int c0   = (t & 31) * 4;
    int s = src[edge];
    int d = dst[edge];
    float inv = 1.0f / fmaxf(cnt_e[d], 1.0f);
    v4bf v = *(const v4bf*)(hb + (size_t)s * DD + c0);
    float* p = msum + (size_t)d * DD + c0;
    atomicAdd(p + 0, (float)v[0] * inv);
    atomicAdd(p + 1, (float)v[1] * inv);
    atomicAdd(p + 2, (float)v[2] * inv);
    atomicAdd(p + 3, (float)v[3] * inv);
}

// ---------------- fused per-layer kernel ----------------
// out[row,:] = sum_e relu?( h@Ws_e + neigh_e@Wn_e + b_e ),  + BN partial sums.
//
// Block: 256 threads = 8 waves; block covers 128 rows x 128 cols.
// Wave w owns rows [blk*128 + w*16, +16) = 8 accumulator tiles of 16x16.
// Weights staged per (etype, matrix) into LDS (34.8KB, row-padded), B-frags via ds_load_b128.
//
// Fragment layouts (ISA 7.12.2, 16-bit):
//  A 16x32: lane m<16 holds row m, K={k0..k0+7, k0+16..k0+23}; lane m+16: K={k0+8..15, k0+24..31}
//  B 32x16: lane n<16 holds col n, K={k0..k0+7, k0+16..k0+23}; lane n+16: K={k0+8..15, k0+24..31}
//  C/D: VGPR v -> lanes 0-15 row v, lanes 16-31 row v+8; col = lane&15
__global__ __launch_bounds__(256, 1)
void k_hsage_layer(const bf16* __restrict__ hb,
                   const float* __restrict__ ms0, const float* __restrict__ ms1,
                   const float* __restrict__ ms2,
                   const bf16* __restrict__ WtS_l, const bf16* __restrict__ WtN_l,
                   const float* __restrict__ bias_l,
                   float* __restrict__ outb, float* __restrict__ bnsums,
                   int n_nodes, int relu) {
    __shared__ bf16 ldsW[DD * LDS_STRIDE];   // one 128x128 weight matrix, padded

    const int lane = threadIdx.x & 31;
    const int wave = threadIdx.x >> 5;
    const int row0 = blockIdx.x * 128 + wave * 16;

    int arow = row0 + (lane & 15);
    if (arow >= n_nodes) arow = n_nodes - 1;      // clamp; EXEC stays all-ones for WMMA
    const int kb_a = (lane >> 4) * 8;
    const int kb_b = (lane >> 4) * 8;
    const int bcol = lane & 15;

    const float* msum_e[NT] = {ms0, ms1, ms2};

    v8f accO[8];
#pragma unroll
    for (int t = 0; t < 8; ++t) accO[t] = (v8f){0.f,0.f,0.f,0.f,0.f,0.f,0.f,0.f};

#pragma unroll
    for (int e = 0; e < NT; ++e) {
        v8f acc[8];
#pragma unroll
        for (int t = 0; t < 8; ++t) acc[t] = (v8f){0.f,0.f,0.f,0.f,0.f,0.f,0.f,0.f};

#pragma unroll
        for (int half = 0; half < 2; ++half) {
            // ---- stage weight matrix for this (etype, half) into LDS ----
            const bf16* Wsrc = (half == 0 ? WtS_l : WtN_l) + e * (DD * DD);
            __syncthreads();
#pragma unroll
            for (int i = 0; i < 8; ++i) {
                int idx = threadIdx.x + i * 256;      // 0..2047 chunks of 8 bf16
                int wr  = idx >> 4;                   // weight row (output col)
                int c8  = (idx & 15) * 8;
                *(v8bf*)(&ldsW[wr * LDS_STRIDE + c8]) = *(const v8bf*)(Wsrc + wr * DD + c8);
            }
            __syncthreads();

            // ---- 4 k-steps of 32 over this half ----
#pragma unroll
            for (int ks = 0; ks < 4; ++ks) {
                const int k0 = ks * 32;
                v16bf a;
                if (half == 0) {
                    // A from bf16 h
                    const bf16* ap = hb + (size_t)arow * DD + k0 + kb_a;
                    v8bf alo = *(const v8bf*)(ap);
                    v8bf ahi = *(const v8bf*)(ap + 16);
#pragma unroll
                    for (int i = 0; i < 8; ++i) { a[i] = alo[i]; a[i + 8] = ahi[i]; }
                } else {
                    // A from fp32 message-sum (already mean-scaled), convert in-register
                    const float* fp = msum_e[e] + (size_t)arow * DD + k0 + kb_a;
                    v4f f0 = *(const v4f*)(fp);
                    v4f f1 = *(const v4f*)(fp + 4);
                    v4f f2 = *(const v4f*)(fp + 16);
                    v4f f3 = *(const v4f*)(fp + 20);
#pragma unroll
                    for (int i = 0; i < 4; ++i) {
                        a[i]      = (bf16)f0[i];
                        a[i + 4]  = (bf16)f1[i];
                        a[i + 8]  = (bf16)f2[i];
                        a[i + 12] = (bf16)f3[i];
                    }
                }
#pragma unroll
                for (int t = 0; t < 8; ++t) {
                    const bf16* bp = &ldsW[(t * 16 + bcol) * LDS_STRIDE + k0 + kb_b];
                    v8bf blo = *(const v8bf*)(bp);
                    v8bf bhi = *(const v8bf*)(bp + 16);
                    v16bf b;
#pragma unroll
                    for (int i = 0; i < 8; ++i) { b[i] = blo[i]; b[i + 8] = bhi[i]; }
                    acc[t] = __builtin_amdgcn_wmma_f32_16x16x32_bf16(
                        false, a, false, b, (short)0, acc[t], false, false);
                }
            }
        }

        // ---- per-etype epilogue: bias + optional relu, accumulate into accO ----
        const float* be = bias_l + e * DD;
#pragma unroll
        for (int t = 0; t < 8; ++t) {
            const float bv = be[t * 16 + bcol];
#pragma unroll
            for (int v = 0; v < 8; ++v) {
                float o = acc[t][v] + bv;
                if (relu) o = fmaxf(o, 0.0f);
                accO[t][v] += o;
            }
        }
    }

    // ---- store (single write, no RMW) + BN partial sums from registers ----
    const int rbase = row0 + ((lane >> 4) * 8);
#pragma unroll
    for (int t = 0; t < 8; ++t) {
        const int col = t * 16 + bcol;
        float s = 0.f, q = 0.f;
#pragma unroll
        for (int v = 0; v < 8; ++v) {
            const int row = rbase + v;
            if (row < n_nodes) {
                float o = accO[t][v];
                outb[(size_t)row * DD + col] = o;
                s += o;
                q += o * o;
            }
        }
        // combine the two half-wave row groups (lane and lane^16 share `col`)
        s += __shfl_xor(s, 16, 32);
        q += __shfl_xor(q, 16, 32);
        if (lane < 16) {
            atomicAdd(&bnsums[col], s);
            atomicAdd(&bnsums[DD + col], q);
        }
    }
}

// ---------------- BatchNorm apply ----------------
__global__ void k_bn_apply(const float* __restrict__ x, const float* __restrict__ sums,
                           const float* __restrict__ g, const float* __restrict__ bta,
                           bf16* __restrict__ ybf, float* __restrict__ yf32,
                           int write_bf, int n) {
    int i = blockIdx.x * blockDim.x + threadIdx.x;   // over NN*DD
    if (i >= n) return;
    int col = i & 127;
    const float inv_n = 1.0f / (float)NN;
    float mu  = sums[col] * inv_n;
    float var = sums[DD + col] * inv_n - mu * mu;
    float y = g[col] * (x[i] - mu) * rsqrtf(var + BN_EPS) + bta[col];
    if (write_bf) ybf[i] = (bf16)y;
    else          yf32[i] = y;
}

// ---------------- host orchestration ----------------

extern "C" void kernel_launch(void* const* d_in, const int* in_sizes, int n_in,
                              void* d_out, int out_size, void* d_ws, size_t ws_size,
                              hipStream_t stream) {
    (void)in_sizes; (void)n_in; (void)out_size; (void)ws_size;

    const float* feat  = (const float*)d_in[0];
    const int*   src   = (const int*)  d_in[1];   // [NT][NE]
    const int*   dst   = (const int*)  d_in[2];   // [NT][NE]
    const float* Wself = (const float*)d_in[3];   // [NL][NT][D][D]
    const float* Wneig = (const float*)d_in[4];
    const float* bias  = (const float*)d_in[5];   // [NL][NT][D]
    const float* gamma = (const float*)d_in[6];   // [NL][D]
    const float* beta  = (const float*)d_in[7];   // [NL][D]

    // workspace carving (256-byte aligned)
    size_t off = 0;
    auto carve = [&](size_t bytes) -> char* {
        char* p = (char*)d_ws + off;
        off += (bytes + 255) & ~(size_t)255;
        return p;
    };
    bf16*  h_bf   = (bf16*)  carve((size_t)NN * DD * sizeof(bf16));      // 25.6 MB
    float* msum[NT];
    for (int e = 0; e < NT; ++e)
        msum[e]  = (float*) carve((size_t)NN * DD * sizeof(float));      // 3 x 51.2 MB
    float* outb   = (float*) carve((size_t)NN * DD * sizeof(float));     // 51.2 MB
    float* cnt    = (float*) carve((size_t)NT * NN * sizeof(float));     // 1.2 MB
    bf16*  WtS    = (bf16*)  carve((size_t)NL * NT * DD * DD * sizeof(bf16));
    bf16*  WtN    = (bf16*)  carve((size_t)NL * NT * DD * DD * sizeof(bf16));
    float* bnsums = (float*) carve(2 * DD * sizeof(float));

    const int ND = NN * DD;
    const int TB = 256;
    auto blks = [](int n, int tb) { return (n + tb - 1) / tb; };

    // one-time (per call) prep
    {
        int nw = NL * NT * DD * DD;
        k_conv_weights<<<blks(nw, TB), TB, 0, stream>>>(Wself, Wneig, WtS, WtN, nw);
        k_zero_f32<<<blks(NT * NN, TB), TB, 0, stream>>>(cnt, NT * NN);
        k_count<<<blks(NT * NE, TB), TB, 0, stream>>>(dst, cnt, NT * NE);
        k_f32_to_bf16<<<blks(ND, TB), TB, 0, stream>>>(feat, h_bf, ND);
    }

    const int layer_blocks = (NN + 127) / 128;      // 782
    const int scat_threads = NE * 32;

    for (int l = 0; l < NL; ++l) {
        const int relu = (l < NL - 1) ? 1 : 0;
        // mean aggregation per etype (scatter pre-scaled by 1/deg)
        for (int e = 0; e < NT; ++e) {
            k_zero_f32<<<blks(ND, TB), TB, 0, stream>>>(msum[e], ND);
            k_scatter<<<blks(scat_threads, TB), TB, 0, stream>>>(
                h_bf, src + (size_t)e * NE, dst + (size_t)e * NE,
                cnt + (size_t)e * NN, msum[e], scat_threads);
        }
        k_zero_f32<<<1, 2 * DD, 0, stream>>>(bnsums, 2 * DD);
        // fused: 3-etype dual-GEMM + bias + relu + sum + out write + BN partials
        k_hsage_layer<<<layer_blocks, TB, 0, stream>>>(
            h_bf, msum[0], msum[1], msum[2],
            WtS + (size_t)l * NT * DD * DD,
            WtN + (size_t)l * NT * DD * DD,
            bias + (size_t)l * NT * DD,
            outb, bnsums, NN, relu);
        // batchnorm apply: bf16 h for next layer, fp32 d_out for the last
        const int last = (l == NL - 1);
        k_bn_apply<<<blks(ND, TB), TB, 0, stream>>>(
            outb, bnsums, gamma + (size_t)l * DD, beta + (size_t)l * DD,
            h_bf, (float*)d_out, last ? 0 : 1, ND);
    }
}
